// CACISLoss_91104846282973
// MI455X (gfx1250) — compile-verified, matrix operand order checked by reading, not run on hardware
//
#include <hip/hip_runtime.h>
#include <hip/hip_bf16.h>
#include <stdint.h>

typedef __attribute__((ext_vector_type(16))) _Float16 v16h;
typedef __attribute__((ext_vector_type(8)))  float    v8f;
typedef int gv4i __attribute__((vector_size(16)));   // matches builtin's int4 param

#define AS1 __attribute__((address_space(1)))
#define AS3 __attribute__((address_space(3)))

#define KDIM    128
#define MSTRIDE 129          // pad M rows -> conflict-free column reads
#define BLK     256
#define FW_ITER 50

__global__ __launch_bounds__(BLK, 1)
void cacis_main(const float* __restrict__ scores,
                const float* __restrict__ C,
                const int*   __restrict__ targets,
                float* __restrict__ wraw,
                float* __restrict__ wratio)
{
    __shared__ __align__(16) float Cs[KDIM * KDIM];      // 64 KB
    __shared__ __align__(16) float Ms[KDIM * MSTRIDE];   // 64.5 KB
    __shared__ float fsc[2][KDIM];     // [0]=scores[b], [1]=base scores
    __shared__ float colsum[KDIM];
    __shared__ float grad[KDIM];
    __shared__ float alpha[KDIM];
    __shared__ float red [BLK];
    __shared__ float red2[BLK];
    __shared__ int   redi[BLK];
    __shared__ float sc[8];

    const int b = blockIdx.x;
    const int t = threadIdx.x;
    const float* Cb = C + (size_t)b * (KDIM * KDIM);

    // ---- stage C[b] (64 KB) into LDS: async-to-LDS path when available ----
#if defined(__gfx1250__) && __has_builtin(__builtin_amdgcn_global_load_async_to_lds_b128)
    #pragma unroll
    for (int i = 0; i < 16; ++i) {
        const int idx4 = i * BLK + t;                       // float4 granule
        __builtin_amdgcn_global_load_async_to_lds_b128(
            (AS1 gv4i*)(AS1 void*)(Cb + idx4 * 4),
            (AS3 gv4i*)(AS3 void*)(&Cs[idx4 * 4]),
            0, 0);
    }
  #if __has_builtin(__builtin_amdgcn_s_wait_asynccnt)
    __builtin_amdgcn_s_wait_asynccnt(0);
  #else
    asm volatile("s_wait_asynccnt 0" ::: "memory");
  #endif
#else
    for (int i = 0; i < 16; ++i) {
        const int idx4 = i * BLK + t;
        ((float4*)Cs)[idx4] = ((const float4*)Cb)[idx4];
    }
#endif
    if (t < KDIM) fsc[0][t] = scores[(size_t)b * KDIM + t];
    __syncthreads();

    // ---- column sums -> base scores; bank-friendly (j varies across lanes) ----
    {
        const int j = t & (KDIM - 1);
        const int h = t >> 7;
        float ps = 0.f;
        const int i0 = h * 64;
        for (int i = i0; i < i0 + 64; ++i) ps += Cs[i * KDIM + j];
        red[t] = ps;
        __syncthreads();
        if (t < KDIM) {
            const float cs = red[t] + red[t + KDIM];
            colsum[t] = cs;
            fsc[1][t] = -cs * (1.0f / KDIM);
        }
        __syncthreads();
    }

    // ---- total sum + trace -> eps ----
    {
        red [t] = (t < KDIM) ? colsum[t]          : 0.f;
        red2[t] = (t < KDIM) ? Cs[t * (KDIM + 1)] : 0.f;
        __syncthreads();
        for (int s = BLK / 2; s >= 1; s >>= 1) {
            if (t < s) { red[t] += red[t + s]; red2[t] += red2[t + s]; }
            __syncthreads();
        }
        if (t == 0) {
            const float off = red[0] - red2[0];
            const float e   = off / (float)(KDIM * KDIM - KDIM);
            sc[0] = fmaxf(e, 1e-8f);
        }
        __syncthreads();
    }
    const float eps = sc[0];
    const int   tb  = targets[b];

    for (int p = 0; p < 2; ++p) {
        const float* fv = fsc[p];

        // ---- gmin = min_ij (0.5 f_i + 0.5 f_j + C_ij)  (shift = -gmin/eps) ----
        const int irow = t >> 1;
        const int j0   = (t & 1) * 64;
        const int rot  = (irow + 32 * (t & 1)) & 63;     // de-conflict LDS banks
        {
            const float hi = 0.5f * fv[irow];
            float lmin = 3.4e38f;
            for (int jj = 0; jj < 64; ++jj) {
                const int j = j0 + ((jj + rot) & 63);
                const float g = hi + 0.5f * fv[j] + Cs[irow * KDIM + j];
                lmin = fminf(lmin, g);
            }
            red[t] = lmin;
            __syncthreads();
            for (int s = BLK / 2; s >= 1; s >>= 1) {
                if (t < s) red[t] = fminf(red[t], red[t + s]);
                __syncthreads();
            }
        }
        const float gmin    = red[0];
        const float inv_eps = 1.0f / eps;
        __syncthreads();

        // ---- M_ij = exp(-(g_ij - gmin)/eps)  in (0,1], stored padded ----
        {
            const float hi = 0.5f * fv[irow];
            for (int jj = 0; jj < 64; ++jj) {
                const int j = j0 + ((jj + rot) & 63);
                const float g = hi + 0.5f * fv[j] + Cs[irow * KDIM + j];
                Ms[irow * MSTRIDE + j] = __expf(-(g - gmin) * inv_eps);
            }
        }
        if (t < KDIM) alpha[t] = 1.0f / KDIM;
        __syncthreads();

        // ---- grad0 = (2/K) * (M · 1) via v_wmma_f32_16x16x32_f16 ----
        {
            const int w    = t >> 5;            // wave -> 16-row block
            const int lane = t & 31;
            const int row  = 16 * w + (lane & 15);
            const int koff = (lane >= 16) ? 8 : 0;
            v16h bone;
            #pragma unroll
            for (int e = 0; e < 16; ++e) bone[e] = (_Float16)1.0f;
            v8f acc = {0.f, 0.f, 0.f, 0.f, 0.f, 0.f, 0.f, 0.f};
            #pragma unroll
            for (int kc = 0; kc < 4; ++kc) {
                v16h a;
                const float* rp = &Ms[row * MSTRIDE + kc * 32 + koff];
                #pragma unroll
                for (int v = 0; v < 4; ++v) {           // K = koff+0..7
                    a[2 * v]     = (_Float16)rp[2 * v];
                    a[2 * v + 1] = (_Float16)rp[2 * v + 1];
                }
                #pragma unroll
                for (int v = 4; v < 8; ++v) {           // K = koff+16..23
                    a[2 * v]     = (_Float16)rp[16 + 2 * (v - 4)];
                    a[2 * v + 1] = (_Float16)rp[16 + 2 * (v - 4) + 1];
                }
                acc = __builtin_amdgcn_wmma_f32_16x16x32_f16(
                          false, a, false, bone, (short)0, acc, false, false);
            }
            // D columns are identical (B = ones): lane0 holds rows w*16+0..7,
            // lane16 holds rows w*16+8..15 (one row per C/D VGPR).
            if (lane == 0) {
                #pragma unroll
                for (int v = 0; v < 8; ++v) grad[16 * w + v] = (2.0f / KDIM) * acc[v];
            } else if (lane == 16) {
                #pragma unroll
                for (int v = 0; v < 8; ++v) grad[16 * w + 8 + v] = (2.0f / KDIM) * acc[v];
            }
        }
        __syncthreads();

        // ---- Frank-Wolfe with rank-1 gradient updates ----
        for (int it = 0; it < FW_ITER; ++it) {
            if (t < KDIM) { red[t] = grad[t]; redi[t] = t; }
            __syncthreads();
            for (int s = KDIM / 2; s >= 1; s >>= 1) {
                if (t < s) {
                    if (red[t + s] < red[t]) { red[t] = red[t + s]; redi[t] = redi[t + s]; }
                }
                __syncthreads();
            }
            const int   imin = redi[0];
            const float step = 2.0f / (float)(it + 2);
            if (t < KDIM) {
                const float col = Ms[t * MSTRIDE + imin];   // conflict-free (pad 129)
                grad[t]  = (1.0f - step) * grad[t]  + 2.0f * step * col;
                alpha[t] = (1.0f - step) * alpha[t] + ((t == imin) ? step : 0.f);
            }
            __syncthreads();
        }

        // ---- q = alpha^T M alpha ;  conj = -eps*log(q) + gmin ----
        {
            const int j = t & (KDIM - 1);
            const int h = t >> 7;
            float vpart = 0.f;
            const int i0 = h * 64;
            for (int i = i0; i < i0 + 64; ++i) vpart += alpha[i] * Ms[i * MSTRIDE + j];
            red[t] = alpha[j] * vpart;
            __syncthreads();
            for (int s = BLK / 2; s >= 1; s >>= 1) {
                if (t < s) red[t] += red[t + s];
                __syncthreads();
            }
            if (t == 0) {
                const float q = red[0];
                sc[2 + p] = (-eps * __logf(q) + gmin) - fv[tb];
            }
            __syncthreads();
        }
    }

    if (t == 0) {
        const float raw  = sc[2];
        const float base = sc[3];
        wraw[b]   = raw;
        wratio[b] = raw / base;
    }
}

__global__ __launch_bounds__(BLK, 1)
void cacis_reduce(const float* __restrict__ wraw,
                  const float* __restrict__ wratio,
                  float* __restrict__ out, int n)
{
    __shared__ float s1[BLK], s2[BLK];
    const int t = threadIdx.x;
    float a = 0.f, c = 0.f;
    for (int i = t; i < n; i += BLK) { a += wraw[i]; c += wratio[i]; }
    s1[t] = a; s2[t] = c;
    __syncthreads();
    for (int s = BLK / 2; s >= 1; s >>= 1) {
        if (t < s) { s1[t] += s1[t + s]; s2[t] += s2[t + s]; }
        __syncthreads();
    }
    if (t == 0) {
        out[0] = s1[0] / (float)n;   // loss = mean(raw)
        out[1] = s2[0] / (float)n;   // loss_norm = mean(raw/base)
    }
}

extern "C" void kernel_launch(void* const* d_in, const int* in_sizes, int n_in,
                              void* d_out, int out_size, void* d_ws, size_t ws_size,
                              hipStream_t stream)
{
    const float* scores  = (const float*)d_in[0];
    const float* C       = (const float*)d_in[1];
    const int*   targets = (const int*)d_in[2];
    const int B = in_sizes[2];                 // 4096 batch elements

    float* wraw   = (float*)d_ws;
    float* wratio = wraw + B;

    cacis_main<<<B, BLK, 0, stream>>>(scores, C, targets, wraw, wratio);
    cacis_reduce<<<1, BLK, 0, stream>>>(wraw, wratio, (float*)d_out, B);
}
